// CausalSelfAttention_35424890257429
// MI455X (gfx1250) — matrix operand order repaired and checked by reading, not compiled, not run on hardware
//
#include <hip/hip_runtime.h>
#include <hip/hip_bf16.h>

// ---------------------------------------------------------------------------
// Problem constants (from reference): BS=4, SL=2048, NE=1024, NH=16, HD=64
// ---------------------------------------------------------------------------
#define BS 4
#define SL 2048
#define NE 1024
#define NH 16
#define HD 64
#define MROWS (BS * SL)          // 8192

typedef __attribute__((ext_vector_type(16))) __bf16          v16bf;
typedef __attribute__((ext_vector_type(16))) unsigned short  v16u;
typedef __attribute__((ext_vector_type(8)))  float           v8f;
typedef __attribute__((ext_vector_type(4)))  unsigned int    v4u;
typedef __attribute__((ext_vector_type(8)))  unsigned int    v8u;

#if defined(__gfx1250__)
#define USE_CDNA5_ASM 1
#else
#define USE_CDNA5_ASM 0
#endif

// float -> bf16 with round-to-nearest-even (software fallback)
static __device__ __forceinline__ unsigned short f2bf(float f) {
    unsigned int u = __float_as_uint(f);
    u = u + 0x7FFFu + ((u >> 16) & 1u);
    return (unsigned short)(u >> 16);
}

// pack two floats into two bf16 (v_cvt_pk_bf16_f32 when available)
static __device__ __forceinline__ unsigned int pack2bf(float lo, float hi) {
#if __has_builtin(__builtin_amdgcn_cvt_pk_bf16_f32)
    typedef __attribute__((ext_vector_type(2))) __bf16 v2bf;
    v2bf t = __builtin_amdgcn_cvt_pk_bf16_f32(lo, hi);
    return __builtin_bit_cast(unsigned int, t);
#else
    return (unsigned int)f2bf(lo) | ((unsigned int)f2bf(hi) << 16);
#endif
}

static __device__ __forceinline__ v8f wmma_bf16(v16u a, v16u b, v8f c) {
    return __builtin_amdgcn_wmma_f32_16x16x32_bf16(
        false, __builtin_bit_cast(v16bf, a),
        false, __builtin_bit_cast(v16bf, b),
        (short)0, c, false, false);
}

#if USE_CDNA5_ASM
// byte offset of a generic pointer to __shared__ data within the LDS aperture
typedef __attribute__((address_space(3))) const void* lds_cptr_t;
static __device__ __forceinline__ unsigned int lds_offset(const void* p) {
    return (unsigned int)(unsigned long long)(lds_cptr_t)p;
}
#endif

// A-fragment (16 x 32 bf16, M x K) from LDS laid out [row][k] (row stride ldr
// ushorts).  Layout: lanes 0-15 row M=lane hold K 0..15 (v0-7); lanes 16-31
// hold K 16..31 (matches load_b_frag).
static __device__ __forceinline__ v16u
load_a_frag(const unsigned short* base, int ldr, int lane, int kbase) {
    int row = lane & 15;
    int kh  = (lane >> 4) & 1;
    const unsigned short* p = base + row * ldr + kbase;
    v16u a;
#pragma unroll
    for (int i = 0; i < 4; ++i) {
        unsigned int w0 = *(const unsigned int*)(p + kh * 16 + 2 * i);
        a[2 * i]     = (unsigned short)(w0 & 0xffffu);
        a[2 * i + 1] = (unsigned short)(w0 >> 16);
        unsigned int w1 = *(const unsigned int*)(p + kh * 16 + 8 + 2 * i);
        a[8 + 2 * i]     = (unsigned short)(w1 & 0xffffu);
        a[8 + 2 * i + 1] = (unsigned short)(w1 >> 16);
    }
    return a;
}

// B-fragment (32 x 16 bf16, K x N) from LDS laid out [n][k] (row stride ldr).
// Contiguous k order: lanes 0-15 col N=lane hold K 0..15 (v0-7); lanes 16-31 K 16..31.
static __device__ __forceinline__ v16u
load_b_frag(const unsigned short* base, int ldr, int lane, int kbase) {
    int n  = lane & 15;
    int kh = (lane >> 4) & 1;
    const unsigned short* p = base + n * ldr + kbase + kh * 16;
    v16u b;
#pragma unroll
    for (int i = 0; i < 8; ++i) {
        unsigned int w = *(const unsigned int*)(p + 2 * i);
        b[2 * i]     = (unsigned short)(w & 0xffffu);
        b[2 * i + 1] = (unsigned short)(w >> 16);
    }
    return b;
}

// ---------------------------------------------------------------------------
// WMMA GEMM: C[M,N] = A[M,K] * B[K,N] + bias.
// MODE 0: A = fp32 x, B = fp32 Wqkv -> scatter bf16 into Q/K/V [B,NH,SL,HD],
//         folding the 1/sqrt(HD) attention scale into Q.
// MODE 1: A = bf16 Y (workspace), B = fp32 Wo -> fp32 output.
// Block: 256 threads = 8 waves as 2x4, block tile 128x128, K-step 32.
// ---------------------------------------------------------------------------
template <int MODE>
__global__ __launch_bounds__(256) void gemm_wmma_kernel(
    const float* __restrict__ Af, const unsigned short* __restrict__ Abf,
    const float* __restrict__ B, const float* __restrict__ bias,
    unsigned short* __restrict__ Qo, unsigned short* __restrict__ Ko,
    unsigned short* __restrict__ Vo, float* __restrict__ Co,
    int N, int Kd) {
    constexpr int BM = 128, BN = 128, BK = 32, PAD = 8;
    __shared__ __align__(16) unsigned short As[BM][BK + PAD];   // [m][k]
    __shared__ __align__(16) unsigned short BsT[BN][BK + PAD];  // [n][k]

    const int tid  = threadIdx.x;
    const int lane = tid & 31;
    const int wave = tid >> 5;
    const int wm   = wave & 1;   // 2 wave-rows
    const int wn   = wave >> 1;  // 4 wave-cols
    const int m0   = blockIdx.x * BM;
    const int n0   = blockIdx.y * BN;

    v8f acc[4][2];
    v8f zero = {};
#pragma unroll
    for (int mt = 0; mt < 4; ++mt)
#pragma unroll
        for (int nt = 0; nt < 2; ++nt) acc[mt][nt] = zero;

    for (int k0 = 0; k0 < Kd; k0 += BK) {
        __syncthreads();
        if (MODE == 0) {
            // fp32 A tile -> bf16 LDS (packed converts)
#pragma unroll
            for (int t = tid; t < BM * BK / 4; t += 256) {
                int r = t >> 3, c = (t & 7) * 4;
                float4 v = *(const float4*)(Af + (size_t)(m0 + r) * Kd + k0 + c);
                *(unsigned int*)&As[r][c]     = pack2bf(v.x, v.y);
                *(unsigned int*)&As[r][c + 2] = pack2bf(v.z, v.w);
            }
        } else {
            // bf16 A tile straight copy
#pragma unroll
            for (int t = tid; t < BM * BK / 2; t += 256) {
                int r = t >> 4, c = (t & 15) * 2;
                *(unsigned int*)&As[r][c] =
                    *(const unsigned int*)(Abf + (size_t)(m0 + r) * Kd + k0 + c);
            }
        }
        // fp32 B tile -> bf16 LDS, transposed to [n][k]
#pragma unroll
        for (int t = tid; t < BK * BN / 4; t += 256) {
            int r = t >> 5, c = (t & 31) * 4;
            float4 v = *(const float4*)(B + (size_t)(k0 + r) * N + n0 + c);
            BsT[c + 0][r] = f2bf(v.x);
            BsT[c + 1][r] = f2bf(v.y);
            BsT[c + 2][r] = f2bf(v.z);
            BsT[c + 3][r] = f2bf(v.w);
        }
        __syncthreads();

        v16u bfrag[2];
#pragma unroll
        for (int nt = 0; nt < 2; ++nt)
            bfrag[nt] = load_b_frag(&BsT[wn * 32 + nt * 16][0], BK + PAD, lane, 0);
#pragma unroll
        for (int mt = 0; mt < 4; ++mt) {
            v16u a = load_a_frag(&As[wm * 64 + mt * 16][0], BK + PAD, lane, 0);
#pragma unroll
            for (int nt = 0; nt < 2; ++nt)
                acc[mt][nt] = wmma_bf16(a, bfrag[nt], acc[mt][nt]);
        }
    }

    // Epilogue. C layout: col = lane&15 (within 16-tile), row = (lane>>4)*8 + r.
    const int col  = lane & 15;
    const int half = lane >> 4;
#pragma unroll
    for (int mt = 0; mt < 4; ++mt) {
#pragma unroll
        for (int nt = 0; nt < 2; ++nt) {
            int gn   = n0 + wn * 32 + nt * 16 + col;
            float bv = bias[gn];
#pragma unroll
            for (int r = 0; r < 8; ++r) {
                int gm  = m0 + wm * 64 + mt * 16 + half * 8 + r;
                float v = acc[mt][nt][r] + bv;
                if (MODE == 0) {
                    int sel = gn >> 10;        // 0=Q 1=K 2=V
                    int c1  = gn & (NE - 1);
                    int h   = c1 >> 6, d = c1 & (HD - 1);
                    int b   = gm >> 11, s = gm & (SL - 1);
                    size_t off = (((size_t)b * NH + h) * SL + s) * HD + d;
                    unsigned short bw = f2bf(sel == 0 ? v * 0.125f : v);
                    unsigned short* dst = (sel == 0) ? Qo : (sel == 1) ? Ko : Vo;
                    dst[off] = bw;
                } else {
                    Co[(size_t)gm * N + gn] = v;
                }
            }
        }
    }
}

// ---------------------------------------------------------------------------
// Flash attention: one block (4 waves / 128 threads) per (b*h, 64-query tile).
// Q pre-scaled by 1/sqrt(HD).  Each wave owns 16 query rows; online softmax
// over causal key tiles; S and O built with bf16 WMMA.
// K tile staged via TDM (tensor_load_to_lds), Q tile via async global->LDS.
// ---------------------------------------------------------------------------
__global__ __launch_bounds__(128) void flash_attn_kernel(
    const unsigned short* __restrict__ Q, const unsigned short* __restrict__ K,
    const unsigned short* __restrict__ V, unsigned short* __restrict__ Y) {
    constexpr int PAD = 8, LD = HD + PAD;
    __shared__ __align__(16) unsigned short Qs[64][LD];     // [qrow][hd]
    __shared__ __align__(16) unsigned short Ks[64][LD];     // [key][hd]  (B of QK^T)
    __shared__ __align__(16) unsigned short VsT[HD][64 + PAD]; // [hd][key] (B of PV)
    __shared__ __align__(16) unsigned short Ps[4][16][64 + PAD]; // per-wave P (A of PV)

    const int tid  = threadIdx.x;
    const int lane = tid & 31;
    const int wave = tid >> 5;
    const int bh   = blockIdx.y;
    const int b    = bh >> 4;      // NH = 16
    const int h    = bh & (NH - 1);
    const int qt   = blockIdx.x;
    const int q0   = qt * 64;

    const size_t head_base = (size_t)bh * SL * HD;
    const unsigned short* Qp = Q + head_base + (size_t)q0 * HD;

    // ---- Stage Q tile (async global->LDS on gfx1250) ----
#if USE_CDNA5_ASM
    {
        const unsigned int qs_base = lds_offset(&Qs[0][0]);
        for (int t = tid; t < 64 * HD / 8; t += 128) {
            int r = t >> 3, c = (t & 7) * 8;
            unsigned int dst = qs_base + (unsigned int)(r * (LD * 2) + c * 2);
            unsigned long long src = (unsigned long long)(size_t)(Qp + r * HD + c);
            asm volatile("global_load_async_to_lds_b128 %0, %1, off"
                         :: "v"(dst), "v"(src) : "memory");
        }
        asm volatile("s_wait_asynccnt 0x0" ::: "memory");
    }
#else
    for (int t = tid; t < 64 * HD / 8; t += 128) {
        int r = t >> 3, c = (t & 7) * 8;
        *(uint4*)&Qs[r][c] = *(const uint4*)(Qp + r * HD + c);
    }
#endif
    __syncthreads();

    // Q A-fragments held in registers across the whole key loop
    v16u aq[2];
    aq[0] = load_a_frag(&Qs[wave * 16][0], LD, lane, 0);
    aq[1] = load_a_frag(&Qs[wave * 16][0], LD, lane, 32);

    v8f o[4];
    v8f zero = {};
#pragma unroll
    for (int nt = 0; nt < 4; ++nt) o[nt] = zero;
    float run_m[8], run_l[8];
#pragma unroll
    for (int r = 0; r < 8; ++r) { run_m[r] = -1e30f; run_l[r] = 0.0f; }

    const int col  = lane & 15;
    const int half = lane >> 4;

#if USE_CDNA5_ASM
    const unsigned int ks_base = lds_offset(&Ks[0][0]);
#endif

    for (int kt = 0; kt <= qt; ++kt) {
        __syncthreads();
        const unsigned short* Kp = K + head_base + (size_t)kt * 64 * HD;
        const unsigned short* Vp = V + head_base + (size_t)kt * 64 * HD;

#if USE_CDNA5_ASM
        // ---- K tile via Tensor Data Mover: 2-D 64x64 bf16 tile, LDS rows
        // padded by 4 dwords (pad_enable, interval 32 dwords, amount 4). ----
        if (wave == 0) {
            unsigned long long ga = (unsigned long long)(size_t)Kp;
            v4u g0;
            g0[0] = 1u;                                       // count=1
            g0[1] = ks_base;                                  // lds_addr (bytes)
            g0[2] = (unsigned int)ga;                         // global_addr[31:0]
            g0[3] = (unsigned int)(ga >> 32) | (2u << 30);    // addr[56:32] | type=2
            v8u g1;
            g1[0] = (1u << 16)        // data_size = 1 (2 bytes)
                  | (1u << 20)        // pad_enable
                  | (4u << 22)        // pad_interval: 32 dwords (one 64-elem row)
                  | (3u << 25);       // pad_amount: 4 dwords (8 ushorts)
            g1[1] = (unsigned int)HD << 16;   // tensor_dim0[15:0] in bits[63:48]
            g1[2] = 64u << 16;                // tensor_dim1[15:0] in bits[95:80]
            g1[3] = (unsigned int)HD << 16;   // tile_dim0 in bits[127:112]
            g1[4] = 64u;                      // tile_dim1=64, tile_dim2=0
            g1[5] = (unsigned int)HD;         // tensor_dim0_stride[31:0]
            g1[6] = 0u;
            g1[7] = 0u;
            asm volatile("tensor_load_to_lds %0, %1" :: "s"(g0), "s"(g1) : "memory");
        }
#else
        for (int t = tid; t < 64 * HD / 8; t += 128) {
            int r = t >> 3, c = (t & 7) * 8;
            *(uint4*)&Ks[r][c] = *(const uint4*)(Kp + r * HD + c);
        }
#endif
        // ---- V tile staged transposed [hd][key] (manual: needs transpose) ----
        for (int t = tid; t < 64 * HD / 8; t += 128) {
            int r = t >> 3, c = (t & 7) * 8;
            uint4 vv = *(const uint4*)(Vp + r * HD + c);
            VsT[c + 0][r] = (unsigned short)(vv.x & 0xffffu);
            VsT[c + 1][r] = (unsigned short)(vv.x >> 16);
            VsT[c + 2][r] = (unsigned short)(vv.y & 0xffffu);
            VsT[c + 3][r] = (unsigned short)(vv.y >> 16);
            VsT[c + 4][r] = (unsigned short)(vv.z & 0xffffu);
            VsT[c + 5][r] = (unsigned short)(vv.z >> 16);
            VsT[c + 6][r] = (unsigned short)(vv.w & 0xffffu);
            VsT[c + 7][r] = (unsigned short)(vv.w >> 16);
        }
        if (kt < qt) {  // global_prefetch_b8 for next tile
            __builtin_prefetch(Kp + 64 * HD, 0, 1);
            __builtin_prefetch(Vp + 64 * HD, 0, 1);
        }
#if USE_CDNA5_ASM
        if (wave == 0) {
#if __has_builtin(__builtin_amdgcn_s_wait_tensorcnt)
            __builtin_amdgcn_s_wait_tensorcnt(0);
#else
            asm volatile("s_wait_tensorcnt 0x0" ::: "memory");
#endif
        }
#endif
        __syncthreads();

        // S = Q * K^T  (16 x 64 per wave)
        v8f s[4];
#pragma unroll
        for (int nt = 0; nt < 4; ++nt) s[nt] = zero;
#pragma unroll
        for (int ks = 0; ks < 2; ++ks) {
#pragma unroll
            for (int nt = 0; nt < 4; ++nt) {
                v16u bk = load_b_frag(&Ks[nt * 16][0], LD, lane, ks * 32);
                s[nt] = wmma_bf16(aq[ks], bk, s[nt]);
            }
        }

        // Causal mask: only the diagonal tile is partially masked
        if (kt == qt) {
#pragma unroll
            for (int nt = 0; nt < 4; ++nt) {
                int kg = nt * 16 + col;
#pragma unroll
                for (int r = 0; r < 8; ++r)
                    if (kg > wave * 16 + half * 8 + r) s[nt][r] = -1e30f;
            }
        }

        // Online softmax per row (row = half*8 + r, spread over 16 lanes)
#pragma unroll
        for (int r = 0; r < 8; ++r) {
            float mx = fmaxf(fmaxf(s[0][r], s[1][r]), fmaxf(s[2][r], s[3][r]));
#pragma unroll
            for (int off = 1; off < 16; off <<= 1)
                mx = fmaxf(mx, __shfl_xor(mx, off, 32));
            float mnew  = fmaxf(run_m[r], mx);
            float alpha = __expf(run_m[r] - mnew);
            float rs = 0.0f;
#pragma unroll
            for (int nt = 0; nt < 4; ++nt) {
                float p = __expf(s[nt][r] - mnew);
                s[nt][r] = p;
                rs += p;
            }
#pragma unroll
            for (int off = 1; off < 16; off <<= 1)
                rs += __shfl_xor(rs, off, 32);
            run_l[r] = run_l[r] * alpha + rs;
            run_m[r] = mnew;
#pragma unroll
            for (int nt = 0; nt < 4; ++nt) o[nt][r] *= alpha;
        }

        // P (C-layout regs) -> LDS bf16 so it can be re-read in A-frag layout
#pragma unroll
        for (int nt = 0; nt < 4; ++nt)
#pragma unroll
            for (int r = 0; r < 8; ++r)
                Ps[wave][half * 8 + r][nt * 16 + col] = f2bf(s[nt][r]);

        // O += P * V
#pragma unroll
        for (int ks = 0; ks < 2; ++ks) {
            v16u ap = load_a_frag(&Ps[wave][0][0], 64 + PAD, lane, ks * 32);
#pragma unroll
            for (int nt = 0; nt < 4; ++nt) {
                v16u bv = load_b_frag(&VsT[nt * 16][0], 64 + PAD, lane, ks * 32);
                o[nt] = wmma_bf16(ap, bv, o[nt]);
            }
        }
    }

    // Epilogue: normalize and store bf16 Y [B*SL, NE]
#pragma unroll
    for (int nt = 0; nt < 4; ++nt) {
#pragma unroll
        for (int r = 0; r < 8; ++r) {
            int qg = q0 + wave * 16 + half * 8 + r;
            float val = o[nt][r] / run_l[r];
            Y[((size_t)b * SL + qg) * NE + h * HD + nt * 16 + col] = f2bf(val);
        }
    }
}

// ---------------------------------------------------------------------------
// Host launcher
// ---------------------------------------------------------------------------
extern "C" void kernel_launch(void* const* d_in, const int* in_sizes, int n_in,
                              void* d_out, int out_size, void* d_ws, size_t ws_size,
                              hipStream_t stream) {
    (void)in_sizes; (void)n_in; (void)out_size; (void)ws_size;
    const float* x    = (const float*)d_in[0];
    // d_in[1] = causal mask (computed analytically in-kernel)
    const float* Wqkv = (const float*)d_in[2];
    const float* bqkv = (const float*)d_in[3];
    const float* Wo   = (const float*)d_in[4];
    const float* bo   = (const float*)d_in[5];
    float* out = (float*)d_out;

    const size_t headElems = (size_t)BS * NH * SL * HD;  // 8M bf16 each
    unsigned short* Qb = (unsigned short*)d_ws;
    unsigned short* Kb = Qb + headElems;
    unsigned short* Vb = Kb + headElems;
    unsigned short* Yb = Vb + headElems;                 // [B*SL, NE] bf16

    // 1) QKV projection: [8192,1024] x [1024,3072] + bias -> Q/K/V bf16
    dim3 g1(MROWS / 128, (3 * NE) / 128);
    gemm_wmma_kernel<0><<<g1, 256, 0, stream>>>(x, nullptr, Wqkv, bqkv,
                                                Qb, Kb, Vb, nullptr,
                                                3 * NE, NE);

    // 2) Causal flash attention per (b*h, 64-row query tile)
    dim3 g2(SL / 64, BS * NH);
    flash_attn_kernel<<<g2, 128, 0, stream>>>(Qb, Kb, Vb, Yb);

    // 3) Output projection: [8192,1024] x [1024,1024] + bias -> fp32 out
    dim3 g3(MROWS / 128, NE / 128);
    gemm_wmma_kernel<1><<<g3, 256, 0, stream>>>(nullptr, Yb, Wo, bo,
                                                nullptr, nullptr, nullptr, out,
                                                NE, NE);
}
